// Block_32109175504962
// MI455X (gfx1250) — matrix operand order repaired and checked by reading, not compiled, not run on hardware
//
#include <hip/hip_runtime.h>
#include <hip/hip_bf16.h>
#include <math.h>

// ---------------- shapes (fixed by the reference) ----------------
#define Bn   8
#define Nn   3136          // 56*56 tokens
#define Cc   256
#define NHh  8
#define HDd  32
#define LLl  9
#define Hh   56
#define Ww   56
#define SRr  8
#define PHp  7
#define PLl  49            // 7*7 pooled tokens
#define PLp  64            // K-padded pooled length (zero fill)
#define Pp   (Bn*PLl)      // 392
#define Mm   (Bn*Nn)       // 25088 tokens
#define HFf  682
#define HFp  704           // K-padded hidden (zero fill, 22*32, 16B-aligned rows)
#define F1o  (2*HFf)       // 1364
#define Tt   1024

typedef __attribute__((ext_vector_type(16))) __bf16 v16bf;
typedef __attribute__((ext_vector_type(8)))  __bf16 v8bf;
typedef __attribute__((ext_vector_type(8)))  float  v8f;

#define PACK16(lo, hi) __builtin_shufflevector(lo, hi, 0,1,2,3,4,5,6,7,8,9,10,11,12,13,14,15)

__device__ __forceinline__ float gelu_exact(float v) {
    return 0.5f * v * (1.f + erff(v * 0.70710678118654752f));
}
__device__ __forceinline__ float softplusf(float t) {
    return (t > 20.f) ? t : log1pf(expf(t));
}

// ---------------- generic WMMA GEMM (bf16 in, fp32 accum) ----------------
// D[m,n] = sum_k A[m,k] * Bw[n,k]  (+bias[n]) (+epilogue) ; batched over z=(b,h)
// epilogue: 0 = bias only, 1 = bias + exact GELU, 2 = bias + residual add
// Each wave computes a 64x16 tile (4 M-subtiles reuse one B fragment).
// Fast path: 10x global_load_b128 + 4x v_wmma per 32-K step, double-buffered.
__global__ __launch_bounds__(256)
void wmma_gemm_kernel(const __bf16* __restrict__ A, long long ldA, long long aB, long long aH,
                      const __bf16* __restrict__ Bw, long long ldB, long long bB, long long bH,
                      float* Df, __bf16* Dbf, long long ldD, long long dB, long long dH,
                      const float* __restrict__ bias, const float* __restrict__ resid,
                      int M, int Ncol, int K, int nhDiv, int epilogue)
{
    const int lane = threadIdx.x & 31;
    const int wave = threadIdx.x >> 5;
    const int m0 = blockIdx.x * 64;
    const int n0 = blockIdx.y * 128 + wave * 16;
    if (m0 >= M || n0 >= Ncol) return;           // wave-uniform exit (EXEC stays full)

    const int bb = blockIdx.z / nhDiv;
    const int hh = blockIdx.z % nhDiv;
    const __bf16* Ab = A  + (long long)bb * aB + (long long)hh * aH;
    const __bf16* Bb = Bw + (long long)bb * bB + (long long)hh * bH;
    const long long baseD = (long long)bb * dB + (long long)hh * dH;

    const int half  = lane >> 4;
    const int rbase = m0 + (lane & 15);          // A row of subtile 0 for this lane
    const int col   = n0 + (lane & 15);          // B row (= output col) for this lane
    const bool colOk = (col < Ncol);

    const __bf16* Ar = Ab + (long long)rbase * ldA;
    const __bf16* Br = Bb + (long long)col * ldB;

    v8f c[4];
#pragma unroll
    for (int r = 0; r < 4; ++r)
#pragma unroll
        for (int i = 0; i < 8; ++i) c[r][i] = 0.f;

    // wave-uniform fast-path eligibility
    const bool fastOK = ((ldA & 7) == 0) && ((ldB & 7) == 0) &&
                        (m0 + 64 <= M) && (n0 + 16 <= Ncol);
    const int Kfast  = fastOK ? (K & ~31) : 0;
    const int nsteps = Kfast >> 5;

    if (nsteps > 0) {
        // A layout: lanes 0-15 K=[0..8)+[16..24), lanes 16-31 K=[8..16)+[24..32)
        // B layout: same as A, lanes 0-15 K=[0..8)+[16..24), lanes 16-31 K=[8..16)+[24..32)
        v8bf al0[4], ah0[4], bl0, bh0;
        bl0 = *(const v8bf*)(Br + half * 8);
        bh0 = *(const v8bf*)(Br + 16 + half * 8);
#pragma unroll
        for (int r = 0; r < 4; ++r) {
            const __bf16* Arr = Ar + (long long)(r * 16) * ldA;
            al0[r] = *(const v8bf*)(Arr + half * 8);
            ah0[r] = *(const v8bf*)(Arr + 16 + half * 8);
        }
#pragma unroll 2
        for (int s = 1; s < nsteps; ++s) {
            const int k0 = s << 5;
            v8bf al1[4], ah1[4], bl1, bh1;        // prefetch next step before computing
            bl1 = *(const v8bf*)(Br + k0 + half * 8);
            bh1 = *(const v8bf*)(Br + k0 + 16 + half * 8);
#pragma unroll
            for (int r = 0; r < 4; ++r) {
                const __bf16* Arr = Ar + (long long)(r * 16) * ldA;
                al1[r] = *(const v8bf*)(Arr + k0 + half * 8);
                ah1[r] = *(const v8bf*)(Arr + k0 + 16 + half * 8);
            }
            v16bf bfrag = PACK16(bl0, bh0);
#pragma unroll
            for (int r = 0; r < 4; ++r) {
                v16bf afrag = PACK16(al0[r], ah0[r]);
                c[r] = __builtin_amdgcn_wmma_f32_16x16x32_bf16(false, afrag, false, bfrag,
                                                               (short)0, c[r], false, false);
            }
#pragma unroll
            for (int r = 0; r < 4; ++r) { al0[r] = al1[r]; ah0[r] = ah1[r]; }
            bl0 = bl1; bh0 = bh1;
        }
        v16bf bfrag = PACK16(bl0, bh0);
#pragma unroll
        for (int r = 0; r < 4; ++r) {
            v16bf afrag = PACK16(al0[r], ah0[r]);
            c[r] = __builtin_amdgcn_wmma_f32_16x16x32_bf16(false, afrag, false, bfrag,
                                                           (short)0, c[r], false, false);
        }
    }

    // guarded remainder / edge-tile path
    for (int k0 = Kfast; k0 < K; k0 += 32) {
        v16bf bfr;
#pragma unroll
        for (int v = 0; v < 8; ++v) {
            int k = k0 + half * 8 + (v >> 2) * 16 + 2 * (v & 3);
            bfr[2*v]     = (colOk && k     < K) ? Br[k]     : (__bf16)0.f;
            bfr[2*v + 1] = (colOk && k + 1 < K) ? Br[k + 1] : (__bf16)0.f;
        }
#pragma unroll
        for (int r = 0; r < 4; ++r) {
            const int row = rbase + r * 16;
            const bool rowOk = (row < M);
            const __bf16* Arr = Ar + (long long)(r * 16) * ldA;
            v16bf afr;
#pragma unroll
            for (int j = 0; j < 4; ++j) {
                int k  = k0 + half * 8 + 2 * j;
                afr[2*j]     = (rowOk && k     < K) ? Arr[k]     : (__bf16)0.f;
                afr[2*j + 1] = (rowOk && k + 1 < K) ? Arr[k + 1] : (__bf16)0.f;
                int k2 = k0 + 16 + half * 8 + 2 * j;
                afr[8 + 2*j]     = (rowOk && k2     < K) ? Arr[k2]     : (__bf16)0.f;
                afr[8 + 2*j + 1] = (rowOk && k2 + 1 < K) ? Arr[k2 + 1] : (__bf16)0.f;
            }
            c[r] = __builtin_amdgcn_wmma_f32_16x16x32_bf16(false, afr, false, bfr,
                                                           (short)0, c[r], false, false);
        }
    }

    // C/D layout: VGPR r -> M = base + r + 8*half ; N = n0 + lane%16
#pragma unroll
    for (int r16 = 0; r16 < 4; ++r16) {
#pragma unroll
        for (int r = 0; r < 8; ++r) {
            int rr = m0 + r16 * 16 + r + 8 * half;
            if (rr < M && colOk) {
                float v = c[r16][r];
                if (bias) v += bias[col];
                if (epilogue == 1) v = gelu_exact(v);
                long long idx = baseD + (long long)rr * ldD + col;
                if (epilogue == 2 && resid) v += resid[idx];
                if (Df)  Df[idx]  = v;
                if (Dbf) Dbf[idx] = (__bf16)v;
            }
        }
    }
}

// ---------------- fp32 -> bf16 cast ----------------
__global__ void cast_bf16_kernel(const float* __restrict__ src, __bf16* __restrict__ dst, int n) {
    int i = blockIdx.x * blockDim.x + threadIdx.x;
    if (i < n) dst[i] = (__bf16)src[i];
}

// fp32 [rows,K] -> bf16 [rows,Kpad] with zero fill (keeps fast-path alignment)
__global__ void cast_pad_bf16_kernel(const float* __restrict__ src, __bf16* __restrict__ dst,
                                     int rows, int K, int Kpad) {
    long long gid = (long long)blockIdx.x * blockDim.x + threadIdx.x;
    if (gid >= (long long)rows * Kpad) return;
    int c = (int)(gid % Kpad);
    int r = (int)(gid / Kpad);
    dst[gid] = (c < K) ? (__bf16)src[(long long)r * K + c] : (__bf16)0.f;
}

// ---------------- LayerNorm over C=256, one row per block ----------------
__global__ __launch_bounds__(256)
void ln_kernel(const float* __restrict__ x, const float* __restrict__ w, const float* __restrict__ b,
               float* of, __bf16* ob, int rows)
{
    int row = blockIdx.x;
    if (row >= rows) return;
    int tid = threadIdx.x;
    long long base = (long long)row * Cc;
    float v = x[base + tid];
    __shared__ float sh[256];
    sh[tid] = v; __syncthreads();
    for (int s = 128; s > 0; s >>= 1) { if (tid < s) sh[tid] += sh[tid + s]; __syncthreads(); }
    float mean = sh[0] / (float)Cc;
    __syncthreads();
    float d = v - mean;
    sh[tid] = d * d; __syncthreads();
    for (int s = 128; s > 0; s >>= 1) { if (tid < s) sh[tid] += sh[tid + s]; __syncthreads(); }
    float var = sh[0] / (float)Cc;
    float o = d * rsqrtf(var + 1e-5f) * w[tid] + b[tid];
    if (of) of[base + tid] = o;
    if (ob) ob[base + tid] = (__bf16)o;
}

// ---------------- q: l2norm + query_embedding + softplus(temp)*seq_scale ----------------
__global__ void qscale_kernel(const float* __restrict__ q, const float* __restrict__ qe,
                              const float* __restrict__ temp, const float* __restrict__ seq,
                              float* __restrict__ qn, float* __restrict__ qs, __bf16* __restrict__ qsb)
{
    int gid = blockIdx.x * blockDim.x + threadIdx.x;
    if (gid >= Mm * NHh) return;
    int m = gid / NHh, h = gid % NHh;
    long long base = (long long)m * Cc + h * HDd;
    float s = 0.f, vals[HDd];
#pragma unroll
    for (int d = 0; d < HDd; ++d) { vals[d] = q[base + d]; s += vals[d] * vals[d]; }
    float inv = 1.f / fmaxf(sqrtf(s), 1e-12f);
    float scale = softplusf(temp[h]) * seq[0];
#pragma unroll
    for (int d = 0; d < HDd; ++d) {
        float n = vals[d] * inv;
        qn[base + d] = n;
        float sc = (n + qe[h * HDd + d]) * scale;
        qs[base + d] = sc;
        qsb[base + d] = (__bf16)sc;
    }
}

// ---------------- k_l l2norm in place (kv cols 0..255) ----------------
__global__ void klnorm_kernel(float* __restrict__ kv)
{
    int gid = blockIdx.x * blockDim.x + threadIdx.x;
    if (gid >= Mm * NHh) return;
    int m = gid / NHh, h = gid % NHh;
    long long base = (long long)m * (2 * Cc) + h * HDd;
    float s = 0.f;
#pragma unroll
    for (int d = 0; d < HDd; ++d) { float v = kv[base + d]; s += v * v; }
    float inv = 1.f / fmaxf(sqrtf(s), 1e-12f);
#pragma unroll
    for (int d = 0; d < HDd; ++d) kv[base + d] *= inv;
}

// ---------------- local 3x3 window attention logits ----------------
__global__ void attn_local_kernel(const float* __restrict__ qs, const float* __restrict__ kv,
                                  const float* __restrict__ rpb, const unsigned char* __restrict__ mask,
                                  float* __restrict__ attn)
{
    int gid = blockIdx.x * blockDim.x + threadIdx.x;
    if (gid >= Mm * NHh) return;
    int m = gid / NHh, h = gid % NHh;
    int b = m / Nn, n = m % Nn;
    int y = n / Ww, x = n % Ww;
    float qv[HDd];
    long long qb = (long long)m * Cc + h * HDd;
#pragma unroll
    for (int d = 0; d < HDd; ++d) qv[d] = qs[qb + d];
    long long rowoff = (long long)(((long long)b * NHh + h) * Nn + n) * (LLl + PLl);
    for (int iy = 0; iy < 3; ++iy)
        for (int jx = 0; jx < 3; ++jx) {
            int l = iy * 3 + jx;
            int yy = y + iy - 1, xx = x + jx - 1;
            float s = 0.f;
            if (yy >= 0 && yy < Hh && xx >= 0 && xx < Ww) {
                long long kb = ((long long)b * Nn + yy * Ww + xx) * (2 * Cc) + h * HDd;
#pragma unroll
                for (int d = 0; d < HDd; ++d) s += qv[d] * kv[kb + d];
            }
            s += rpb[h * LLl + l];
            if (mask[n * LLl + l]) s = -INFINITY;
            attn[rowoff + l] = s;
        }
}

// ---------------- 8x8 average pooling of the GELU(sr) map ----------------
__global__ void pool_kernel(const float* __restrict__ g, float* __restrict__ pooled)
{
    int gid = blockIdx.x * blockDim.x + threadIdx.x;
    if (gid >= Bn * PLl * Cc) return;
    int c = gid % Cc, p = (gid / Cc) % PLl, b = gid / (Cc * PLl);
    int py = p / PHp, px = p % PHp;
    float s = 0.f;
    for (int dy = 0; dy < SRr; ++dy)
        for (int dx = 0; dx < SRr; ++dx) {
            int n = (py * SRr + dy) * Ww + (px * SRr + dx);
            s += g[((long long)b * Nn + n) * Cc + c];
        }
    pooled[((long long)b * PLl + p) * Cc + c] = s * (1.f / 64.f);
}

// ---------------- k_pool l2norm -> bf16, v_pool transpose -> bf16 (K-padded to 64) ----------------
__global__ void kpoolprep_kernel(const float* __restrict__ kvp2, __bf16* __restrict__ kp, __bf16* __restrict__ vpt)
{
    int gid = blockIdx.x * blockDim.x + threadIdx.x;
    if (gid >= Bn * NHh * PLp) return;
    int p = gid % PLp, bh = gid / PLp;
    int b = bh / NHh, h = bh % NHh;
    if (p >= PLl) {                       // zero-fill v_pool^T pad columns
#pragma unroll
        for (int d = 0; d < HDd; ++d)
            vpt[((long long)bh * HDd + d) * PLp + p] = (__bf16)0.f;
        return;
    }
    long long base = ((long long)b * PLl + p) * (2 * Cc);
    float kx[HDd], s = 0.f;
#pragma unroll
    for (int d = 0; d < HDd; ++d) { kx[d] = kvp2[base + h * HDd + d]; s += kx[d] * kx[d]; }
    float inv = 1.f / fmaxf(sqrtf(s), 1e-12f);
#pragma unroll
    for (int d = 0; d < HDd; ++d) {
        kp[((long long)bh * PLl + p) * HDd + d] = (__bf16)(kx[d] * inv);
        vpt[((long long)bh * HDd + d) * PLp + p] = (__bf16)kvp2[base + (NHh + h) * HDd + d];
    }
}

// ---------------- continuous position bias MLP: relu(tbl@w1^T+b1)@w2^T+b2 ----------------
__global__ __launch_bounds__(256)
void cpb_kernel(const float* __restrict__ tbl, const float* __restrict__ w1, const float* __restrict__ b1,
                const float* __restrict__ w2, const float* __restrict__ b2, float* __restrict__ cpb)
{
    int t = blockIdx.x, tid = threadIdx.x;
    float x0 = tbl[t * 2 + 0], x1 = tbl[t * 2 + 1];
    float part[NHh];
#pragma unroll
    for (int h = 0; h < NHh; ++h) part[h] = 0.f;
    for (int jj = 0; jj < 2; ++jj) {
        int j = tid + jj * 256;
        float hd = fmaxf(0.f, x0 * w1[j * 2 + 0] + x1 * w1[j * 2 + 1] + b1[j]);
#pragma unroll
        for (int h = 0; h < NHh; ++h) part[h] += hd * w2[h * 512 + j];
    }
    __shared__ float red[NHh][256];
#pragma unroll
    for (int h = 0; h < NHh; ++h) red[h][tid] = part[h];
    __syncthreads();
    for (int s = 128; s > 0; s >>= 1) {
        if (tid < s)
#pragma unroll
            for (int h = 0; h < NHh; ++h) red[h][tid] += red[h][tid + s];
        __syncthreads();
    }
    if (tid < NHh) cpb[t * NHh + tid] = red[tid][0] + b2[tid];
}

// ---------------- softmax over [local(9) | pool(49)+cpb-gather]; a_pool K-padded ----------------
__global__ void softmax_kernel(float* __restrict__ attn, const float* __restrict__ cpb,
                               const int* __restrict__ rpi, __bf16* __restrict__ apool)
{
    int gid = blockIdx.x * blockDim.x + threadIdx.x;
    if (gid >= Bn * NHh * Nn) return;
    int bh = gid / Nn, n = gid % Nn, h = bh % NHh;
    long long rowoff = (long long)gid * (LLl + PLl);
    float vals[LLl + PLl];
#pragma unroll
    for (int l = 0; l < LLl; ++l) vals[l] = attn[rowoff + l];
    for (int j = 0; j < PLl; ++j)
        vals[LLl + j] = attn[rowoff + LLl + j] + cpb[rpi[n * PLl + j] * NHh + h];
    float mx = -INFINITY;
    for (int i = 0; i < LLl + PLl; ++i) mx = fmaxf(mx, vals[i]);
    float sum = 0.f;
    for (int i = 0; i < LLl + PLl; ++i) { float e = expf(vals[i] - mx); vals[i] = e; sum += e; }
    float inv = 1.f / sum;
#pragma unroll
    for (int l = 0; l < LLl; ++l) attn[rowoff + l] = vals[l] * inv;        // a_loc (f32)
    long long ab = (long long)gid * PLp;
    for (int j = 0; j < PLl; ++j)
        apool[ab + j] = (__bf16)(vals[LLl + j] * inv);                     // a_pool (bf16)
    for (int j = PLl; j < PLp; ++j) apool[ab + j] = (__bf16)0.f;           // K pad
}

// ---------------- x_local: (q_norm·learnable + bias + a_loc) against windowed v ----------------
__global__ void xlocal_kernel(const float* __restrict__ qn, const float* __restrict__ ltw,
                              const float* __restrict__ lb, const float* __restrict__ attn,
                              const float* __restrict__ kv, float* __restrict__ ao, __bf16* __restrict__ aob)
{
    int gid = blockIdx.x * blockDim.x + threadIdx.x;
    if (gid >= Mm * NHh) return;
    int m = gid / NHh, h = gid % NHh;
    int b = m / Nn, n = m % Nn;
    int y = n / Ww, x = n % Ww;
    float q[HDd];
    long long qb = (long long)m * Cc + h * HDd;
#pragma unroll
    for (int d = 0; d < HDd; ++d) q[d] = qn[qb + d];
    long long rowoff = (long long)(((long long)b * NHh + h) * Nn + n) * (LLl + PLl);
    float coeff[LLl];
#pragma unroll
    for (int l = 0; l < LLl; ++l) {
        float s = 0.f;
#pragma unroll
        for (int d = 0; d < HDd; ++d) s += q[d] * ltw[h * (HDd * LLl) + d * LLl + l];
        coeff[l] = s + lb[h * LLl + l] + attn[rowoff + l];
    }
    float acc[HDd];
#pragma unroll
    for (int d = 0; d < HDd; ++d) acc[d] = 0.f;
    for (int iy = 0; iy < 3; ++iy)
        for (int jx = 0; jx < 3; ++jx) {
            int yy = y + iy - 1, xx = x + jx - 1;
            if (yy < 0 || yy >= Hh || xx < 0 || xx >= Ww) continue;
            float cf = coeff[iy * 3 + jx];
            long long vb = ((long long)b * Nn + yy * Ww + xx) * (2 * Cc) + Cc + h * HDd;
#pragma unroll
            for (int d = 0; d < HDd; ++d) acc[d] += cf * kv[vb + d];
        }
#pragma unroll
    for (int d = 0; d < HDd; ++d) {
        long long idx = (long long)m * Cc + h * HDd + d;
        float t = ao[idx] + acc[d];            // ao already holds x_pool from the WMMA GEMM
        ao[idx] = t;
        aob[idx] = (__bf16)t;
    }
}

// ---------------- depthwise 3x3 + bias + GELU gate (writes K-padded hg) ----------------
__global__ void dwgate_kernel(const __bf16* __restrict__ h, const float* __restrict__ dww,
                              const float* __restrict__ dwb, __bf16* __restrict__ hg)
{
    long long gid = (long long)blockIdx.x * blockDim.x + threadIdx.x;
    if (gid >= (long long)Mm * HFp) return;
    int c = (int)(gid % HFp);
    int m = (int)(gid / HFp);
    if (c >= HFf) { hg[gid] = (__bf16)0.f; return; }   // K pad
    int b = m / Nn, n = m % Nn;
    int y = n / Ww, x = n % Ww;
    float s = 0.f;
    for (int ky = 0; ky < 3; ++ky)
        for (int kx = 0; kx < 3; ++kx) {
            int yy = y + ky - 1, xx = x + kx - 1;
            if (yy < 0 || yy >= Hh || xx < 0 || xx >= Ww) continue;
            float in = (float)h[((long long)b * Nn + yy * Ww + xx) * F1o + c];
            s += dww[c * 9 + ky * 3 + kx] * in;
        }
    s += dwb[c];
    float gate = (float)h[(long long)m * F1o + HFf + c];
    hg[gid] = (__bf16)(gelu_exact(s) * gate);
}

// ================= launch =================
extern "C" void kernel_launch(void* const* d_in, const int* in_sizes, int n_in,
                              void* d_out, int out_size, void* d_ws, size_t ws_size,
                              hipStream_t stream)
{
    const float* x        = (const float*)d_in[0];
    const float* tbl      = (const float*)d_in[1];
    const float* seq      = (const float*)d_in[2];
    const float* n1w      = (const float*)d_in[3];
    const float* n1b      = (const float*)d_in[4];
    const float* q_w      = (const float*)d_in[5];
    const float* q_b      = (const float*)d_in[6];
    const float* kv_w     = (const float*)d_in[7];
    const float* kv_b     = (const float*)d_in[8];
    const float* temp     = (const float*)d_in[9];
    const float* qe       = (const float*)d_in[10];
    const float* rpb      = (const float*)d_in[11];
    const float* ltw      = (const float*)d_in[12];
    const float* lb       = (const float*)d_in[13];
    const float* sr_w     = (const float*)d_in[14];
    const float* sr_b     = (const float*)d_in[15];
    const float* npw      = (const float*)d_in[16];
    const float* npb      = (const float*)d_in[17];
    const float* c1w      = (const float*)d_in[18];
    const float* c1b      = (const float*)d_in[19];
    const float* c2w      = (const float*)d_in[20];
    const float* c2b      = (const float*)d_in[21];
    const float* proj_w   = (const float*)d_in[22];
    const float* proj_b   = (const float*)d_in[23];
    const float* n2w      = (const float*)d_in[24];
    const float* n2b      = (const float*)d_in[25];
    const float* fc1_w    = (const float*)d_in[26];
    const float* fc1_b    = (const float*)d_in[27];
    const float* dw_w     = (const float*)d_in[28];
    const float* dw_b     = (const float*)d_in[29];
    const float* fc2_w    = (const float*)d_in[30];
    const float* fc2_b    = (const float*)d_in[31];
    const int*   rpi      = (const int*)d_in[32];
    const unsigned char* pmask = (const unsigned char*)d_in[33];
    float* out = (float*)d_out;

    // bump allocator over workspace
    char* wp = (char*)d_ws;
    auto alloc = [&](size_t bytes) -> void* {
        void* p = (void*)wp;
        wp += (bytes + 255) & ~(size_t)255;
        return p;
    };
    auto fbuf = [&](size_t n) { return (float*)alloc(n * sizeof(float)); };
    auto bbuf = [&](size_t n) { return (__bf16*)alloc(n * sizeof(__bf16)); };

    // bf16 weights
    __bf16* qw_b   = bbuf((size_t)Cc * Cc);
    __bf16* kvw_b  = bbuf((size_t)2 * Cc * Cc);
    __bf16* srw_b  = bbuf((size_t)Cc * Cc);
    __bf16* pjw_b  = bbuf((size_t)Cc * Cc);
    __bf16* f1w_b  = bbuf((size_t)F1o * Cc);
    __bf16* f2w_b  = bbuf((size_t)Cc * HFp);            // K-padded
    // activations
    float*  xn_f   = fbuf((size_t)Mm * Cc);
    __bf16* xn_b   = bbuf((size_t)Mm * Cc);
    float*  q_f    = fbuf((size_t)Mm * Cc);
    float*  qn_f   = fbuf((size_t)Mm * Cc);
    float*  qs_f   = fbuf((size_t)Mm * Cc);
    __bf16* qs_b   = bbuf((size_t)Mm * Cc);
    float*  kv_f   = fbuf((size_t)Mm * 2 * Cc);
    float*  sg_f   = fbuf((size_t)Mm * Cc);
    float*  pool_f = fbuf((size_t)Pp * Cc);
    __bf16* pool_b = bbuf((size_t)Pp * Cc);
    float*  kvp2_f = fbuf((size_t)Pp * 2 * Cc);
    __bf16* kp_b   = bbuf((size_t)Bn * NHh * PLl * HDd);
    __bf16* vpt_b  = bbuf((size_t)Bn * NHh * HDd * PLp); // K-padded
    float*  cpb_f  = fbuf((size_t)Tt * NHh);
    float*  attn_f = fbuf((size_t)Bn * NHh * Nn * (LLl + PLl));
    __bf16* ap_b   = bbuf((size_t)Bn * NHh * Nn * PLp);  // K-padded
    float*  ao_f   = fbuf((size_t)Mm * Cc);
    __bf16* ao_b   = bbuf((size_t)Mm * Cc);
    float*  x2_f   = fbuf((size_t)Mm * Cc);
    __bf16* xn2_b  = bbuf((size_t)Mm * Cc);
    __bf16* h_b    = bbuf((size_t)Mm * F1o);
    __bf16* hg_b   = bbuf((size_t)Mm * HFp);             // K-padded

    auto cast = [&](const float* s, __bf16* d, int n) {
        cast_bf16_kernel<<<(n + 255) / 256, 256, 0, stream>>>(s, d, n);
    };
    cast(q_w,   qw_b,  Cc * Cc);
    cast(kv_w,  kvw_b, 2 * Cc * Cc);
    cast(sr_w,  srw_b, Cc * Cc);
    cast(proj_w,pjw_b, Cc * Cc);
    cast(fc1_w, f1w_b, F1o * Cc);
    cast_pad_bf16_kernel<<<(Cc * HFp + 255) / 256, 256, 0, stream>>>(fc2_w, f2w_b, Cc, HFf, HFp);

    // LN1
    ln_kernel<<<Mm, 256, 0, stream>>>(x, n1w, n1b, xn_f, xn_b, Mm);

    auto gemm = [&](const __bf16* A, long long ldA, long long aB, long long aH,
                    const __bf16* Bw, long long ldB, long long bB, long long bH,
                    float* Df, __bf16* Dbf, long long ldD, long long dB, long long dH,
                    const float* bias, const float* resid,
                    int M, int Ncol, int K, int nhDiv, int ep, int zcnt) {
        dim3 grid((M + 63) / 64, (Ncol + 127) / 128, zcnt);
        wmma_gemm_kernel<<<grid, 256, 0, stream>>>(A, ldA, aB, aH, Bw, ldB, bB, bH,
                                                   Df, Dbf, ldD, dB, dH, bias, resid,
                                                   M, Ncol, K, nhDiv, ep);
    };

    // q / kv / sr (1x1 conv == token GEMM with exact-GELU epilogue)
    gemm(xn_b, Cc, 0, 0, qw_b,  Cc, 0, 0, q_f,  nullptr, Cc,     0, 0, q_b,  nullptr, Mm, Cc,     Cc, 1, 0, 1);
    gemm(xn_b, Cc, 0, 0, kvw_b, Cc, 0, 0, kv_f, nullptr, 2 * Cc, 0, 0, kv_b, nullptr, Mm, 2 * Cc, Cc, 1, 0, 1);
    gemm(xn_b, Cc, 0, 0, srw_b, Cc, 0, 0, sg_f, nullptr, Cc,     0, 0, sr_b, nullptr, Mm, Cc,     Cc, 1, 1, 1);

    const int tpb = 256;
    qscale_kernel<<<(Mm * NHh + tpb - 1) / tpb, tpb, 0, stream>>>(q_f, qe, temp, seq, qn_f, qs_f, qs_b);
    klnorm_kernel<<<(Mm * NHh + tpb - 1) / tpb, tpb, 0, stream>>>(kv_f);
    attn_local_kernel<<<(Mm * NHh + tpb - 1) / tpb, tpb, 0, stream>>>(qs_f, kv_f, rpb, pmask, attn_f);

    // pooled path: avgpool -> LN -> kv GEMM -> k l2norm / v transpose (padded)
    pool_kernel<<<(Bn * PLl * Cc + tpb - 1) / tpb, tpb, 0, stream>>>(sg_f, pool_f);
    ln_kernel<<<Pp, 256, 0, stream>>>(pool_f, npw, npb, nullptr, pool_b, Pp);
    gemm(pool_b, Cc, 0, 0, kvw_b, Cc, 0, 0, kvp2_f, nullptr, 2 * Cc, 0, 0, kv_b, nullptr, Pp, 2 * Cc, Cc, 1, 0, 1);
    kpoolprep_kernel<<<(Bn * NHh * PLp + tpb - 1) / tpb, tpb, 0, stream>>>(kvp2_f, kp_b, vpt_b);

    // CPB MLP
    cpb_kernel<<<Tt, 256, 0, stream>>>(tbl, c1w, c1b, c2w, c2b, cpb_f);

    // attn_pool = q_scaled @ k_pool^T (per b,h): K=32, writes into attn cols [9..58)
    gemm(qs_b, Cc, (long long)Nn * Cc, HDd,
         kp_b, HDd, (long long)NHh * PLl * HDd, (long long)PLl * HDd,
         attn_f + LLl, nullptr, LLl + PLl, (long long)NHh * Nn * (LLl + PLl), (long long)Nn * (LLl + PLl),
         nullptr, nullptr, Nn, PLl, HDd, NHh, 0, Bn * NHh);

    softmax_kernel<<<(Bn * NHh * Nn + tpb - 1) / tpb, tpb, 0, stream>>>(attn_f, cpb_f, rpi, ap_b);

    // x_pool = a_pool @ v_pool (per b,h): K padded to 64, fully fast path
    gemm(ap_b, PLp, (long long)NHh * Nn * PLp, (long long)Nn * PLp,
         vpt_b, PLp, (long long)NHh * HDd * PLp, (long long)HDd * PLp,
         ao_f, nullptr, Cc, (long long)Nn * Cc, HDd,
         nullptr, nullptr, Nn, HDd, PLp, NHh, 0, Bn * NHh);

    // x_local accumulates onto x_pool, produces bf16 for proj
    xlocal_kernel<<<(Mm * NHh + tpb - 1) / tpb, tpb, 0, stream>>>(qn_f, ltw, lb, attn_f, kv_f, ao_f, ao_b);

    // proj + residual(x)
    gemm(ao_b, Cc, 0, 0, pjw_b, Cc, 0, 0, x2_f, nullptr, Cc, 0, 0, proj_b, x, Mm, Cc, Cc, 1, 2, 1);

    // MLP: LN2 -> fc1 -> dwconv+GELU gate -> fc2 + residual(x2)
    ln_kernel<<<Mm, 256, 0, stream>>>(x2_f, n2w, n2b, nullptr, xn2_b, Mm);
    gemm(xn2_b, Cc, 0, 0, f1w_b, Cc, 0, 0, nullptr, h_b, F1o, 0, 0, fc1_b, nullptr, Mm, F1o, Cc, 1, 0, 1);
    {
        long long total = (long long)Mm * HFp;
        dwgate_kernel<<<(unsigned)((total + tpb - 1) / tpb), tpb, 0, stream>>>(h_b, dw_w, dw_b, hg_b);
    }
    // fc2: K padded to 704, fully fast path; + residual(x2)
    gemm(hg_b, HFp, 0, 0, f2w_b, HFp, 0, 0, out, nullptr, Cc, 0, 0, fc2_b, x2_f, Mm, Cc, HFp, 1, 2, 1);

    (void)in_sizes; (void)n_in; (void)out_size; (void)ws_size;
}
